// UltimateGeoSNN_50130858279655
// MI455X (gfx1250) — compile-verified, matrix operand order checked by reading, not run on hardware
//
#include <hip/hip_runtime.h>
#include <hip/hip_bf16.h>

// ---------------------------------------------------------------------------
// UltimateGeoSNN forward for MI455X (gfx1250, wave32, WMMA)
// Convs are implicit-im2col GEMMs on v_wmma_f32_16x16x32_f16 with
// double-buffered LDS staging; async global->LDS (ASYNCcnt) when available.
// All BatchNorms / biases / pointwise residuals folded into weights once per
// launch (deterministic: re-derived every call inside the graph).
// ---------------------------------------------------------------------------

typedef __attribute__((ext_vector_type(16))) _Float16 v16h;
typedef __attribute__((ext_vector_type(8)))  float    v8f;
typedef int v4i_gcc __attribute__((vector_size(16)));   // matches builtin proto

#if __has_builtin(__builtin_amdgcn_global_load_async_to_lds_b128) && \
    __has_builtin(__builtin_amdgcn_s_wait_asynccnt)
#define HAVE_ASYNC_LDS 1
#else
#define HAVE_ASYNC_LDS 0
#endif

#define BATCH 512
#define CIN0  15
#define TLEN  320
#define NCLS  4
#define VTH1  0.15f
#define VTH2  0.30f
#define VTH3  0.30f
#define EPSBN 1e-5f

// ---- workspace layout (bytes) ----
static constexpr size_t OFF_PAR = 0;                                   // 16384 f32
static constexpr size_t OFF_W1  = 65536;                               // 128*480 f16
static constexpr size_t OFF_W2  = OFF_W1 + (size_t)128*480*2;          // 256*896 f16
static constexpr size_t OFF_SP  = OFF_W2 + (size_t)256*896*2;          // B*T*32 f16
static constexpr size_t OFF_C1  = OFF_SP + (size_t)BATCH*TLEN*32*2;    // B*T*128 f32
static constexpr size_t OFF_S2  = OFF_C1 + (size_t)BATCH*TLEN*128*4;   // B*T*128 f16
static constexpr size_t OFF_C2  = OFF_S2 + (size_t)BATCH*TLEN*128*2;   // B*T*256 f32
static constexpr size_t OFF_PL  = OFF_C2 + (size_t)BATCH*TLEN*256*4;   // B*2560 f32

// sub-offsets inside the f32 param block (units: floats)
#define P_MU   0
#define P_MV   225
#define P_INJB 450
#define P_AEFF 512
#define P_BEFF 992
#define P_B1   1024
#define P_B2   1152
#define P_FCB  1408
#define P_FCW  2048

struct RawParams {
    const float *A, *Wu_w, *Wu_b, *Wv_w, *Wv_b;
    const float *bi_g, *bi_b, *bi_rm, *bi_rv;            // bn_inj
    const float *spc_w, *spc_b;
    const float *bs_g, *bs_b, *bs_rm, *bs_rv;            // bn_sp
    const float *c1_w, *c1_b;
    const float *b1_g, *b1_b, *b1_rm, *b1_rv;            // bn1
    const float *sc1_w, *sc1_b;
    const float *c2_w, *c2_b;
    const float *b2_g, *b2_b, *b2_rm, *b2_rv;            // bn2
    const float *sc2_w, *sc2_b;
    const float *bf_g, *bf_b, *bf_rm, *bf_rv;            // bn_fc
    const float *fc_w, *fc_b;
};

// async copy helper: 16B global -> LDS (falls back to VGPR round-trip)
__device__ __forceinline__ void copy16_g2l(const _Float16* g, _Float16* l)
{
#if HAVE_ASYNC_LDS
    __builtin_amdgcn_global_load_async_to_lds_b128(
        (__attribute__((address_space(1))) v4i_gcc*)const_cast<_Float16*>(g),
        (__attribute__((address_space(3))) v4i_gcc*)l,
        0, 0);
#else
    *(uint4*)l = *(const uint4*)g;
#endif
}

__device__ __forceinline__ void wait_stage()
{
#if HAVE_ASYNC_LDS
    __builtin_amdgcn_s_wait_asynccnt(0);
#endif
    __syncthreads();
}

// ---------------------------------------------------------------------------
// K0: fold every affine stage into effective weights (single block).
// ---------------------------------------------------------------------------
__global__ void snn_prep_kernel(RawParams p, float* par, _Float16* W1, _Float16* W2)
{
    const int tid = threadIdx.x;
    const int nth = blockDim.x;

    // Mu' = scale_inj * (Wu @ A^T), Mv' likewise   (inj[e] = sum_c M[e,c]*x[c])
    for (int i = tid; i < 225; i += nth) {
        const int e = i / 15, c = i % 15;
        const float sc = p.bi_g[e] * rsqrtf(p.bi_rv[e] + EPSBN);
        float su = 0.f, sv = 0.f;
        for (int d = 0; d < 15; ++d) {
            su += p.Wu_w[e*15 + d] * p.A[c*15 + d];
            sv += p.Wv_w[e*15 + d] * p.A[c*15 + d];
        }
        par[P_MU + i] = sc * su;
        par[P_MV + i] = sc * sv;
    }
    for (int e = tid; e < 15; e += nth) {
        const float sc = p.bi_g[e] * rsqrtf(p.bi_rv[e] + EPSBN);
        par[P_INJB + e] = sc * (p.Wu_b[e] + p.Wv_b[e] - p.bi_rm[e]) + p.bi_b[e];
    }
    // sp = bn_sp(spc @ s1 + spc_b)
    for (int i = tid; i < 32*15; i += nth) {
        const int o = i / 15;
        const float sc = p.bs_g[o] * rsqrtf(p.bs_rv[o] + EPSBN);
        par[P_AEFF + i] = sc * p.spc_w[i];
    }
    for (int o = tid; o < 32; o += nth) {
        const float sc = p.bs_g[o] * rsqrtf(p.bs_rv[o] + EPSBN);
        par[P_BEFF + o] = sc * (p.spc_b[o] - p.bs_rm[o]) + p.bs_b[o];
    }
    // c1: K layout k = tap*32 + c, center tap 7 absorbs sc1 (pointwise residual)
    for (int i = tid; i < 128*480; i += nth) {
        const int o = i / 480, k = i % 480, tap = k / 32, c = k % 32;
        const float sc = p.b1_g[o] * rsqrtf(p.b1_rv[o] + EPSBN);
        float v = sc * p.c1_w[(o*32 + c)*15 + tap];
        if (tap == 7) v += p.sc1_w[o*32 + c];
        W1[i] = (_Float16)v;
    }
    for (int o = tid; o < 128; o += nth) {
        const float sc = p.b1_g[o] * rsqrtf(p.b1_rv[o] + EPSBN);
        par[P_B1 + o] = sc * (p.c1_b[o] - p.b1_rm[o]) + p.b1_b[o] + p.sc1_b[o];
    }
    // c2: K layout k = tap*128 + c, center tap 3 absorbs sc2
    for (int i = tid; i < 256*896; i += nth) {
        const int o = i / 896, k = i % 896, tap = k / 128, c = k % 128;
        const float sc = p.b2_g[o] * rsqrtf(p.b2_rv[o] + EPSBN);
        float v = sc * p.c2_w[(o*128 + c)*7 + tap];
        if (tap == 3) v += p.sc2_w[o*128 + c];
        W2[i] = (_Float16)v;
    }
    for (int o = tid; o < 256; o += nth) {
        const float sc = p.b2_g[o] * rsqrtf(p.b2_rv[o] + EPSBN);
        par[P_B2 + o] = sc * (p.c2_b[o] - p.b2_rm[o]) + p.b2_b[o] + p.sc2_b[o];
    }
    // fc through bn_fc
    for (int i = tid; i < 4*2560; i += nth) {
        const int col = i % 2560;
        const float sc = p.bf_g[col] * rsqrtf(p.bf_rv[col] + EPSBN);
        par[P_FCW + i] = p.fc_w[i] * sc;
    }
    if (tid < 4) {
        float acc = p.fc_b[tid];
        for (int i = 0; i < 2560; ++i) {
            const float sc = p.bf_g[i] * rsqrtf(p.bf_rv[i] + EPSBN);
            acc += p.fc_w[tid*2560 + i] * (p.bf_b[i] - p.bf_rm[i] * sc);
        }
        par[P_FCB + tid] = acc;
    }
}

// ---------------------------------------------------------------------------
// K1: fused injection + per-element decay + LIF scan-1 + 15->32 sp projection.
// One wave32 per batch row; channel mixes via lane shuffles.
// ---------------------------------------------------------------------------
__global__ void snn_scan1_sp_kernel(const float* __restrict__ x,
                                    const float* __restrict__ par,
                                    const float* __restrict__ g_curv,
                                    const float* __restrict__ g_tang,
                                    _Float16* __restrict__ sp)
{
    const int b = blockIdx.x;
    const int lane = threadIdx.x;          // 0..31
    const int row = (lane < CIN0) ? lane : 0;

    float mu[15], mv[15], ae[15];
    #pragma unroll
    for (int c = 0; c < 15; ++c) {
        mu[c] = par[P_MU + row*15 + c];
        mv[c] = par[P_MV + row*15 + c];
        ae[c] = par[P_AEFF + lane*15 + c];
    }
    const float injb = par[P_INJB + row];
    const float be   = par[P_BEFF + lane];
    const float gc = g_curv[0], gt = g_tang[0];

    const size_t xbase = ((size_t)b * CIN0 * TLEN) * 4;
    float v = 0.f;

    for (int t = 0; t < TLEN; ++t) {
        float x0 = 0.f, x1 = 0.f, dec = 0.f;
        if (lane < CIN0) {
            const float4 xx = *(const float4*)&x[xbase + ((size_t)lane*TLEN + t)*4];
            x0 = xx.x; x1 = xx.y;
            float tau = 35.0f * __expf(-(gc*xx.z + gt*xx.w));
            tau = fminf(fmaxf(tau, 5.0f), 100.0f);
            dec = __expf(-1.0f / tau);
        }
        float inj = injb;
        #pragma unroll
        for (int c = 0; c < 15; ++c)
            inj += mu[c]*__shfl(x0, c, 32) + mv[c]*__shfl(x1, c, 32);

        v = v * dec + inj;
        float s = (v - VTH1 > 0.f) ? 1.0f : 0.0f;
        v -= s * VTH1;
        if (lane >= CIN0) s = 0.f;

        float spo = be;
        #pragma unroll
        for (int c = 0; c < 15; ++c)
            spo += ae[c]*__shfl(s, c, 32);

        sp[((size_t)(b*TLEN + t))*32 + lane] = (_Float16)spo;
    }
}

// ---------------------------------------------------------------------------
// Implicit-im2col WMMA GEMM for the temporal convs.
//   Y[b][t][MTOT] = W_eff(MTOT x CIN*TAPS) * col(b,t) + bias
// K-steps of 32 align with (tap, 32-channel chunk). 64-column tiles,
// double-buffered LDS (stride-40 padding, bank-conflict free), async
// global->LDS staging overlapped with WMMA when the toolchain supports it.
// Block = MTOT threads: 4 N-waves x (MTOT/128) M-groups; 8 M-subtiles/wave.
// ---------------------------------------------------------------------------
template<int MTOT, int CINL, int TAPS, int PAD>
__global__ void snn_conv_wmma_kernel(const _Float16* __restrict__ W,
                                     const float* __restrict__ bias,
                                     const _Float16* __restrict__ X,
                                     float* __restrict__ Y)
{
    constexpr int KTOT = CINL * TAPS;
    constexpr int NT = 64;
    constexpr int LSTR = 40;                 // halves; 20 dwords -> conflict-free b128
    __shared__ _Float16 lA[2][MTOT * LSTR];
    __shared__ _Float16 lB[2][NT * LSTR];

    const int b = blockIdx.y;
    const int tbase = blockIdx.x * NT;
    const int tid = threadIdx.x;
    const int lane = tid & 31, w = tid >> 5;
    const int wn = w & 3, wm = w >> 2;
    const int n0 = wn * 16;
    const int nl = lane & 15;
    const int kh = (lane >> 4) * 16;         // contiguous 16-K half per lane half (k = 16h + s)

    v8f acc[8] = {};

    auto stage = [&](int k0, int buf) {
        const int tap = k0 / CINL;
        const int c0  = k0 % CINL;
        // A slab: MTOT x 32 halves
        for (int i = tid; i < MTOT * 4; i += blockDim.x) {
            const int m = i >> 2, ch = (i & 3) * 8;
            copy16_g2l(&W[(size_t)m*KTOT + k0 + ch], &lA[buf][m*LSTR + ch]);
        }
        // B slab: 64 t-rows x 32 channels (zero-fill for time padding)
        for (int i = tid; i < NT * 4; i += blockDim.x) {
            const int n = i >> 2, ch = (i & 3) * 8;
            const int t = tbase + n + tap - PAD;
            _Float16* l = &lB[buf][n*LSTR + ch];
            if (t >= 0 && t < TLEN)
                copy16_g2l(&X[((size_t)(b*TLEN + t))*CINL + c0 + ch], l);
            else
                *(uint4*)l = (uint4){0u, 0u, 0u, 0u};
        }
    };

    stage(0, 0);
    wait_stage();

    for (int k0 = 0; k0 < KTOT; k0 += 32) {
        const int cur = (k0 >> 5) & 1;
        if (k0 + 32 < KTOT)
            stage(k0 + 32, cur ^ 1);        // overlaps with WMMA below

        const v16h bf = *(const v16h*)&lB[cur][(n0 + nl)*LSTR + kh];
        #pragma unroll
        for (int mt = 0; mt < 8; ++mt) {
            const v16h af = *(const v16h*)&lA[cur][(wm*128 + mt*16 + nl)*LSTR + kh];
            acc[mt] = __builtin_amdgcn_wmma_f32_16x16x32_f16(
                false, af, false, bf, (short)0, acc[mt], false, false);
        }
        wait_stage();
    }

    // D layout: lanes 0-15 -> M = r, lanes 16-31 -> M = 8+r; N = lane&15
    const int tcol = tbase + n0 + nl;
    const int mhi = 8 * (lane >> 4);
    #pragma unroll
    for (int mt = 0; mt < 8; ++mt) {
        #pragma unroll
        for (int r = 0; r < 8; ++r) {
            const int m = wm*128 + mt*16 + mhi + r;
            Y[((size_t)(b*TLEN + tcol))*MTOT + m] = acc[mt][r] + bias[m];
        }
    }
}

// ---------------------------------------------------------------------------
// K3: LIF scan-2 (scalar decay), spikes to f16 for the next GEMM.
// ---------------------------------------------------------------------------
__global__ void snn_scan2_kernel(const float* __restrict__ c1,
                                 const float* __restrict__ w_v2,
                                 _Float16* __restrict__ s2)
{
    const int b = blockIdx.x, o = threadIdx.x;     // 128 channels
    const float d2 = 1.0f / (1.0f + __expf(-w_v2[0]));
    float v = 0.f;
    for (int t = 0; t < TLEN; ++t) {
        const size_t idx = ((size_t)(b*TLEN + t))*128 + o;
        v = v * d2 + c1[idx];
        const float s = (v - VTH2 > 0.f) ? 1.0f : 0.0f;
        v -= s * VTH2;
        s2[idx] = (_Float16)s;
    }
}

// ---------------------------------------------------------------------------
// K5: LIF scan-3 + AvgPool(32) -> pooled[b][o*10+seg] (channel-major flatten).
// ---------------------------------------------------------------------------
__global__ void snn_scan3_pool_kernel(const float* __restrict__ c2,
                                      const float* __restrict__ w_v3,
                                      float* __restrict__ pooled)
{
    const int b = blockIdx.x, o = threadIdx.x;     // 256 channels
    const float d3 = 1.0f / (1.0f + __expf(-w_v3[0]));
    float v = 0.f;
    for (int seg = 0; seg < TLEN/32; ++seg) {
        float sum = 0.f;
        for (int j = 0; j < 32; ++j) {
            const int t = seg*32 + j;
            v = v * d3 + c2[((size_t)(b*TLEN + t))*256 + o];
            const float s = (v - VTH3 > 0.f) ? 1.0f : 0.0f;
            v -= s * VTH3;
            sum += s;
        }
        pooled[(size_t)b*2560 + o*10 + seg] = sum * (1.0f/32.0f);
    }
}

// ---------------------------------------------------------------------------
// K6: fc (bn_fc folded): out[b][4] = pooled . fcw_eff + fcb_eff
// ---------------------------------------------------------------------------
__global__ void snn_fc_kernel(const float* __restrict__ pooled,
                              const float* __restrict__ par,
                              float* __restrict__ out)
{
    const int b = blockIdx.x, tid = threadIdx.x;
    __shared__ float red[4][256];
    float p0 = 0.f, p1 = 0.f, p2 = 0.f, p3 = 0.f;
    for (int i = tid; i < 2560; i += 256) {
        const float xv = pooled[(size_t)b*2560 + i];
        p0 += xv * par[P_FCW + 0*2560 + i];
        p1 += xv * par[P_FCW + 1*2560 + i];
        p2 += xv * par[P_FCW + 2*2560 + i];
        p3 += xv * par[P_FCW + 3*2560 + i];
    }
    red[0][tid] = p0; red[1][tid] = p1; red[2][tid] = p2; red[3][tid] = p3;
    __syncthreads();
    for (int s = 128; s > 0; s >>= 1) {
        if (tid < s) {
            #pragma unroll
            for (int j = 0; j < 4; ++j) red[j][tid] += red[j][tid + s];
        }
        __syncthreads();
    }
    if (tid < 4) out[b*4 + tid] = red[tid][0] + par[P_FCB + tid];
}

// ---------------------------------------------------------------------------
extern "C" void kernel_launch(void* const* d_in, const int* in_sizes, int n_in,
                              void* d_out, int out_size, void* d_ws, size_t ws_size,
                              hipStream_t stream)
{
    (void)in_sizes; (void)n_in; (void)out_size; (void)ws_size;
    // setup_inputs dict flattened order: x, then params leaves in insertion order
    const float* x = (const float*)d_in[0];
    RawParams p;
    p.A     = (const float*)d_in[1];
    p.Wu_w  = (const float*)d_in[2];  p.Wu_b = (const float*)d_in[3];
    p.Wv_w  = (const float*)d_in[4];  p.Wv_b = (const float*)d_in[5];
    p.bi_g  = (const float*)d_in[6];  p.bi_b = (const float*)d_in[7];
    p.bi_rm = (const float*)d_in[8];  p.bi_rv= (const float*)d_in[9];
    p.spc_w = (const float*)d_in[10]; p.spc_b= (const float*)d_in[11];
    p.bs_g  = (const float*)d_in[12]; p.bs_b = (const float*)d_in[13];
    p.bs_rm = (const float*)d_in[14]; p.bs_rv= (const float*)d_in[15];
    p.c1_w  = (const float*)d_in[16]; p.c1_b = (const float*)d_in[17];
    p.b1_g  = (const float*)d_in[18]; p.b1_b = (const float*)d_in[19];
    p.b1_rm = (const float*)d_in[20]; p.b1_rv= (const float*)d_in[21];
    p.sc1_w = (const float*)d_in[22]; p.sc1_b= (const float*)d_in[23];
    p.c2_w  = (const float*)d_in[24]; p.c2_b = (const float*)d_in[25];
    p.b2_g  = (const float*)d_in[26]; p.b2_b = (const float*)d_in[27];
    p.b2_rm = (const float*)d_in[28]; p.b2_rv= (const float*)d_in[29];
    p.sc2_w = (const float*)d_in[30]; p.sc2_b= (const float*)d_in[31];
    p.bf_g  = (const float*)d_in[32]; p.bf_b = (const float*)d_in[33];
    p.bf_rm = (const float*)d_in[34]; p.bf_rv= (const float*)d_in[35];
    p.fc_w  = (const float*)d_in[36]; p.fc_b = (const float*)d_in[37];
    const float* g_curv = (const float*)d_in[38];
    const float* g_tang = (const float*)d_in[39];
    const float* w_v2   = (const float*)d_in[40];
    const float* w_v3   = (const float*)d_in[41];

    char* ws = (char*)d_ws;
    float*     par = (float*)(ws + OFF_PAR);
    _Float16*  W1  = (_Float16*)(ws + OFF_W1);
    _Float16*  W2  = (_Float16*)(ws + OFF_W2);
    _Float16*  sp  = (_Float16*)(ws + OFF_SP);
    float*     c1  = (float*)(ws + OFF_C1);
    float*     c2  = (float*)(ws + OFF_C2);
    _Float16*  s2  = (_Float16*)(ws + OFF_S2);
    float*     pl  = (float*)(ws + OFF_PL);
    float*     out = (float*)d_out;

    snn_prep_kernel<<<1, 256, 0, stream>>>(p, par, W1, W2);
    snn_scan1_sp_kernel<<<BATCH, 32, 0, stream>>>(x, par, g_curv, g_tang, sp);
    snn_conv_wmma_kernel<128, 32, 15, 7><<<dim3(TLEN/64, BATCH), 128, 0, stream>>>(
        W1, par + P_B1, sp, c1);
    snn_scan2_kernel<<<BATCH, 128, 0, stream>>>(c1, w_v2, s2);
    snn_conv_wmma_kernel<256, 128, 7, 3><<<dim3(TLEN/64, BATCH), 256, 0, stream>>>(
        W2, par + P_B2, s2, c2);
    snn_scan3_pool_kernel<<<BATCH, 256, 0, stream>>>(c2, w_v3, pl);
    snn_fc_kernel<<<BATCH, 256, 0, stream>>>(pl, par, out);
}
